// ScaledDotProductAttention_57483842290084
// MI455X (gfx1250) — compile-verified, matrix operand order checked by reading, not run on hardware
//
#include <hip/hip_runtime.h>
#include <hip/hip_bf16.h>
#include <cstdint>

// ---------------------------------------------------------------------------
// Flash-attention forward, B=2 H=16 N=2048 D=128, fp32 I/O, causal.
// MI455X (gfx1250, wave32):
//   * bf16 WMMA 16x16x32 (8x FLOPs/instr vs f32 WMMA 16x16x4)
//   * K tiles fetched with global_load_async_to_lds_b128, software-pipelined:
//     tile kt+1 streams in (ASYNCcnt) while tile kt computes
//   * one cooperative f32->bf16 conversion pass per block (not per wave)
//   * fragment-swizzled LDS: every WMMA operand is one contiguous 32B read
//     (2x ds_load_b128, bank-conflict-free)
//   * softmax in base-2 domain: 1/sqrt(d)*log2(e) folded into Q conversion,
//     bare v_exp_f32 per element; accumulator rescale as packed f32 vector ops
// ---------------------------------------------------------------------------

typedef __bf16 bf16;
typedef __attribute__((ext_vector_type(16))) __bf16 v16bf;
typedef __attribute__((ext_vector_type(8)))  __bf16 v8bf;
typedef __attribute__((ext_vector_type(8)))  float  v8f;
typedef __attribute__((ext_vector_type(4)))  float  v4f;

#if __has_builtin(__builtin_amdgcn_exp2f)
#define EXP2(x) __builtin_amdgcn_exp2f(x)
#else
#define EXP2(x) __builtin_exp2f(x)
#endif

constexpr int Bc = 2, Hc = 16, Nc = 2048, Dc = 128;
constexpr int BM = 64;          // query rows per block (4 waves x 16)
constexpr int BN = 64;          // key/value rows per tile
constexpr int NTHREADS = 128;   // 4 wave32s

// Swizzled layouts: per row, fragment slots of 16 bf16 (32B) laid out as
// slot(c, half) holding k = c*32 + half*8 + {0..7} and c*32+16+half*8+{0..7}.
constexpr int KSWZ_STRIDE = 128 + 8;  // bf16/row: 272B (16B aligned, bank-shift 4)
constexpr int VSWZ_STRIDE = 64 + 8;   // bf16/row: 144B (16B aligned, bank-shift 36)
constexpr int PSWZ_STRIDE = 64 + 8;

constexpr int STAGE_OFF   = 0;                            // K fp32 async stage
constexpr int STAGE_BYTES = BN * Dc * 4;                  // 32768
constexpr int SMEM_K_OFF  = STAGE_OFF + STAGE_BYTES;      // 32768
constexpr int SMEM_K_BYTES= BN * KSWZ_STRIDE * 2;         // 17408
constexpr int SMEM_V_OFF  = SMEM_K_OFF + SMEM_K_BYTES;    // 50176
constexpr int SMEM_V_BYTES= Dc * VSWZ_STRIDE * 2;         // 18432
constexpr int SMEM_P_OFF  = SMEM_V_OFF + SMEM_V_BYTES;    // 68608
constexpr int SMEM_P_BYTES= 4 * 16 * PSWZ_STRIDE * 2;     // 9216
constexpr int SMEM_BYTES  = SMEM_P_OFF + SMEM_P_BYTES;    // 77824 (<320KB/WGP)

// 1/sqrt(128) * log2(e): WMMA of (scaled Q)·K directly yields exp2 arguments.
constexpr float QSCL = 0.08838834764831844f * 1.44269504088896340736f;

__device__ __forceinline__ void issue_async_k(unsigned long long kaddr,
                                              int kt, int tid) {
    const unsigned gbase = (unsigned)(kt * (BN * Dc * 4));
    #pragma unroll
    for (int i = 0; i < 16; ++i) {
        const unsigned off16 = (unsigned)((tid + i * NTHREADS) * 16);
        unsigned ldsOff = (unsigned)STAGE_OFF + off16;
        unsigned gOff   = gbase + off16;
        asm volatile("global_load_async_to_lds_b128 %0, %1, %2"
                     :: "v"(ldsOff), "v"(gOff), "s"(kaddr) : "memory");
    }
}

__global__ __launch_bounds__(NTHREADS, 1)
void fa_fwd_kernel(const float* __restrict__ Q, const float* __restrict__ K,
                   const float* __restrict__ V, float* __restrict__ O)
{
    extern __shared__ char smem_raw[];
    float* stage = (float*)(smem_raw + STAGE_OFF);
    bf16*  sK    = (bf16*)(smem_raw + SMEM_K_OFF);
    bf16*  sV    = (bf16*)(smem_raw + SMEM_V_OFF);
    bf16*  sP    = (bf16*)(smem_raw + SMEM_P_OFF);

    const int tid  = threadIdx.x;
    const int wave = tid >> 5;
    const int lane = tid & 31;
    const int lo   = lane & 15;   // fragment row/col index
    const int hi   = lane >> 4;   // half-wave (k-split / row-group)

    const int nmblk = Nc / BM;
    const int bh    = blockIdx.x / nmblk;
    const int mblk  = blockIdx.x % nmblk;

    const float* qhead = Q + (size_t)bh * Nc * Dc;
    const float* vhead = V + (size_t)bh * Nc * Dc;
    float*       ohead = O + (size_t)bh * Nc * Dc;
    const unsigned long long kaddr =
        (unsigned long long)(K + (size_t)bh * Nc * Dc);   // uniform -> SGPR pair

    const int m0 = mblk * BM + wave * 16;  // first query row of this wave

    // ---- prologue: start streaming K tile 0 while we set up Q ----
    issue_async_k(kaddr, 0, tid);

    // ---- Q -> (scaled) bf16 A fragments; scale folded in once per block ----
    v16bf qa[4];
    {
        const float* qrow = qhead + (size_t)(m0 + lo) * Dc;
        #pragma unroll
        for (int c = 0; c < 4; ++c) {
            const int k0 = c * 32 + hi * 8;
            v4f a0 = *(const v4f*)(qrow + k0);
            v4f a1 = *(const v4f*)(qrow + k0 + 4);
            v4f b0 = *(const v4f*)(qrow + k0 + 16);
            v4f b1 = *(const v4f*)(qrow + k0 + 20);
            #pragma unroll
            for (int j = 0; j < 4; ++j) {
                qa[c][j]      = (bf16)(a0[j] * QSCL);
                qa[c][4 + j]  = (bf16)(a1[j] * QSCL);
                qa[c][8 + j]  = (bf16)(b0[j] * QSCL);
                qa[c][12 + j] = (bf16)(b1[j] * QSCL);
            }
        }
    }

    v8f oacc[8];
    #pragma unroll
    for (int i = 0; i < 8; ++i)
        #pragma unroll
        for (int j = 0; j < 8; ++j) oacc[i][j] = 0.f;

    float rmax[8], rsum[8];   // rmax in base-2 (log2) domain
    #pragma unroll
    for (int r = 0; r < 8; ++r) { rmax[r] = -__builtin_inff(); rsum[r] = 0.f; }

    for (int kt = 0; kt <= mblk; ++kt) {           // causal: only tiles <= diagonal
        // B1: stage(kt) fully written (own asyncs done + all waves arrived),
        //     and all waves finished reading sK/sV/P of tile kt-1.
        asm volatile("s_wait_asynccnt 0" ::: "memory");
        __syncthreads();

        // ---- cooperative K conversion: stage f32 -> fragment-swizzled bf16 ----
        #pragma unroll
        for (int p = 0; p < 8; ++p) {
            const int o    = tid + p * NTHREADS;    // 1024 x 16B output blocks
            const int kr   = o >> 4;
            const int rem  = o & 15;
            const int c    = rem >> 2;
            const int half = (rem >> 1) & 1;
            const int jh   = rem & 1;
            const float* src = stage + kr * Dc + c * 32 + jh * 16 + half * 8;
            v4f f0 = *(const v4f*)(src);
            v4f f1 = *(const v4f*)(src + 4);
            v8bf o8;
            #pragma unroll
            for (int j = 0; j < 4; ++j) {
                o8[j]     = (bf16)f0[j];
                o8[4 + j] = (bf16)f1[j];
            }
            *(v8bf*)(sK + kr * KSWZ_STRIDE + (c * 2 + half) * 16 + jh * 8) = o8;
        }
        // B2: stage reads done (safe to refill) and sK visible to all waves.
        __syncthreads();

        // ---- software pipeline: start streaming K tile kt+1 now ----
        if (kt < mblk) issue_async_k(kaddr, kt + 1, tid);

        // ---- V tile kt: global fp32 -> bf16, transposed+swizzled into sV ----
        #pragma unroll
        for (int i = 0; i < 16; ++i) {
            const int e    = tid + i * NTHREADS;
            const int row  = e >> 5;                   // key row in tile (= k index)
            const int d0   = (e & 31) * 4;
            v4f v = *(const v4f*)(vhead + (size_t)(kt * BN + row) * Dc + d0);
            const int c    = row >> 5;
            const int r5   = row & 31;
            const int half = (r5 >> 3) & 1;
            const int jp   = (r5 & 7) | ((r5 >> 4) << 3);
            const int voff = (c * 2 + half) * 16 + jp;
            #pragma unroll
            for (int j = 0; j < 4; ++j)
                sV[(d0 + j) * VSWZ_STRIDE + voff] = (bf16)v[j];
        }

        // ---- S2 = log2e/sqrt(d) * Q K^T : ds_load_b128 -> v_wmma only ----
        v8f sfrag[4];
        #pragma unroll
        for (int sn = 0; sn < 4; ++sn) {
            v8f s;
            #pragma unroll
            for (int j = 0; j < 8; ++j) s[j] = 0.f;
            const bf16* krow = sK + (sn * 16 + lo) * KSWZ_STRIDE + hi * 16;
            #pragma unroll
            for (int c = 0; c < 4; ++c) {
                v16bf kb = *(const v16bf*)(krow + c * 32);
                s = __builtin_amdgcn_wmma_f32_16x16x32_bf16(
                        false, qa[c], false, kb, (short)0, s, false, false);
            }
            sfrag[sn] = s;
        }

        // B3: sV(kt) visible before PV consumes it (S phase above needs only sK).
        __syncthreads();

        // ---- online softmax, base-2 domain. C layout: row m = m0 + hi*8 + r ----
        const bool diag = (kt == mblk);
        float tmax[8];
        #pragma unroll
        for (int r = 0; r < 8; ++r) tmax[r] = -__builtin_inff();
        #pragma unroll
        for (int sn = 0; sn < 4; ++sn) {
            const int ncol = kt * BN + sn * 16 + lo;
            #pragma unroll
            for (int r = 0; r < 8; ++r) {
                float s = sfrag[sn][r];
                if (diag) {
                    const int mrow = m0 + hi * 8 + r;
                    s = (ncol > mrow) ? -1.0e30f : s;
                }
                sfrag[sn][r] = s;
                tmax[r] = fmaxf(tmax[r], s);
            }
        }
        #pragma unroll
        for (int r = 0; r < 8; ++r)
            #pragma unroll
            for (int off = 1; off < 16; off <<= 1)
                tmax[r] = fmaxf(tmax[r], __shfl_xor(tmax[r], off, 32));

        float alpha[8], tsum[8];
        #pragma unroll
        for (int r = 0; r < 8; ++r) {
            const float nm = fmaxf(rmax[r], tmax[r]);
            alpha[r] = EXP2(rmax[r] - nm);            // bare v_exp_f32
            rmax[r]  = nm;
            tsum[r]  = 0.f;
        }

        // P = exp2(S2 - max2) -> bf16, fragment-swizzled, wave-private region
        bf16* pbase = sP + wave * 16 * PSWZ_STRIDE;
        #pragma unroll
        for (int sn = 0; sn < 4; ++sn) {
            const int off_sn = ((sn >> 1) * 2 + (lo >> 3)) * 16 + (sn & 1) * 8 + (lo & 7);
            #pragma unroll
            for (int r = 0; r < 8; ++r) {
                const float pv = EXP2(sfrag[sn][r] - rmax[r]);
                tsum[r] += pv;
                pbase[(hi * 8 + r) * PSWZ_STRIDE + off_sn] = (bf16)pv;
            }
        }

        #pragma unroll
        for (int r = 0; r < 8; ++r) {
            #pragma unroll
            for (int off = 1; off < 16; off <<= 1)
                tsum[r] += __shfl_xor(tsum[r], off, 32);
            rsum[r] = rsum[r] * alpha[r] + tsum[r];
        }

        // rescale O accumulators: vector mul -> packed v_pk_mul_f32
        v8f alphav;
        #pragma unroll
        for (int r = 0; r < 8; ++r) alphav[r] = alpha[r];
        #pragma unroll
        for (int dn = 0; dn < 8; ++dn)
            oacc[dn] *= alphav;

        // ---- O += P V : contiguous fragment loads, LDS in-order per wave ----
        #pragma unroll
        for (int c = 0; c < 2; ++c) {
            v16bf pa = *(const v16bf*)(pbase + lo * PSWZ_STRIDE + (c * 2 + hi) * 16);
            #pragma unroll
            for (int dn = 0; dn < 8; ++dn) {
                v16bf vb = *(const v16bf*)(sV + (dn * 16 + lo) * VSWZ_STRIDE
                                              + (c * 2 + hi) * 16);
                oacc[dn] = __builtin_amdgcn_wmma_f32_16x16x32_bf16(
                               false, pa, false, vb, (short)0, oacc[dn], false, false);
            }
        }
    }

    // ---- normalize and write out (fp32) ----
    #pragma unroll
    for (int r = 0; r < 8; ++r) {
        const float inv = 1.0f / rsum[r];
        float* orow = ohead + (size_t)(m0 + hi * 8 + r) * Dc;
        #pragma unroll
        for (int dn = 0; dn < 8; ++dn)
            orow[dn * 16 + lo] = oacc[dn][r] * inv;
    }
}

extern "C" void kernel_launch(void* const* d_in, const int* in_sizes, int n_in,
                              void* d_out, int out_size, void* d_ws, size_t ws_size,
                              hipStream_t stream) {
    (void)in_sizes; (void)n_in; (void)out_size; (void)d_ws; (void)ws_size;
    const float* Q = (const float*)d_in[0];
    const float* K = (const float*)d_in[1];
    const float* V = (const float*)d_in[2];
    float* O = (float*)d_out;   // (B,H,N,D) fp32; mask input handled analytically

    // CDNA5 allows up to 320KB LDS per workgroup; raise the dynamic-LDS cap.
    (void)hipFuncSetAttribute((const void*)fa_fwd_kernel,
                              hipFuncAttributeMaxDynamicSharedMemorySize,
                              SMEM_BYTES);

    dim3 grid(Bc * Hc * (Nc / BM));
    dim3 block(NTHREADS);
    hipLaunchKernelGGL(fa_fwd_kernel, grid, block, SMEM_BYTES, stream, Q, K, V, O);
}
